// RadixLinearAttention_61495341744735
// MI455X (gfx1250) — matrix operand order repaired and checked by reading, not compiled, not run on hardware
//
#include <hip/hip_runtime.h>

#define TT 4096
#define HQ 32
#define HKV 16
#define DK 128
#define DV 128
#define KCONV 4
#define CONV_DIM (HQ*DK + HKV*DK + HKV*DV) /* 8192 */
#define CHUNK 32
#define DVB 16
#define NCHUNK (TT/CHUNK)

typedef __attribute__((ext_vector_type(16))) __bf16 v16bf;
typedef __attribute__((ext_vector_type(8)))  float  v8f;

#if __has_builtin(__builtin_amdgcn_global_load_async_to_lds_b128)
#define HAVE_ASYNC 1
typedef int v4i_ __attribute__((vector_size(16)));
typedef __attribute__((address_space(1))) v4i_* gptr_t;   // global int4*
typedef __attribute__((address_space(3))) v4i_* lptr_t;   // LDS int4*
#else
#define HAVE_ASYNC 0
#endif

#if __has_builtin(__builtin_amdgcn_tensor_load_to_lds)
#define HAVE_TDM 1
typedef unsigned int u32x4 __attribute__((vector_size(16)));
typedef int          i32x4 __attribute__((vector_size(16)));
typedef int          i32x8 __attribute__((vector_size(32)));
typedef __attribute__((address_space(3))) const void* lconst_void;

__device__ __forceinline__ unsigned lds_byte_off(const void* p) {
  return (unsigned)(size_t)(lconst_void)p;     // AS3 ptrtoint -> LDS byte offset
}
// One TDM descriptor: 2D tile (tile_d1 rows x tile_d0 2-byte elems), row stride
// stride_elems (2-byte units), packed contiguously into LDS at ldsoff.
__device__ __forceinline__ void tdm_load_2d(unsigned ldsoff, const void* gaddr,
                                            unsigned tile_d0, unsigned tile_d1,
                                            unsigned stride_elems) {
  unsigned long long ga = (unsigned long long)(size_t)gaddr;
  u32x4 g0 = {0, 0, 0, 0};
  g0[0] = 1u;                                             // count=1 (valid, user)
  g0[1] = ldsoff;                                         // lds_addr (bytes)
  g0[2] = (unsigned)ga;                                   // global_addr[31:0]
  g0[3] = (unsigned)((ga >> 32) & 0x01FFFFFFu) | (2u << 30); // addr[56:32]|type=2
  i32x8 g1 = {0, 0, 0, 0, 0, 0, 0, 0};
  g1[0] = (int)(1u << 16);            // workgroup_mask=0, data_size=1 (2 bytes)
  g1[1] = 0;                          // tensor_dim0 = 2^30 (low 16 bits -> 0)
  g1[2] = (int)0x4000;                // tensor_dim0 hi | tensor_dim1 lo (=0)
  g1[3] = (int)(0x4000u | (tile_d0 << 16));  // tensor_dim1 hi=2^30 | tile_dim0
  g1[4] = (int)(tile_d1 & 0xFFFFu);   // tile_dim1 | tile_dim2=0
  g1[5] = (int)stride_elems;          // tensor_dim0_stride[31:0]
  i32x4 gz = {0, 0, 0, 0};            // groups 2/3 unused (<=2D tensor)
#if __clang_major__ >= 23
  i32x8 gz8 = {0, 0, 0, 0, 0, 0, 0, 0};
  __builtin_amdgcn_tensor_load_to_lds(g0, g1, gz, gz, gz8, 0);
#else
  __builtin_amdgcn_tensor_load_to_lds(g0, g1, gz, gz, 0);
#endif
}
#else
#define HAVE_TDM 0
#endif

#if HAVE_TDM
#define WAIT_LOADS(pending) do { if (pending) __builtin_amdgcn_s_wait_tensorcnt(3); \
                                 else         __builtin_amdgcn_s_wait_tensorcnt(0); } while (0)
#elif HAVE_ASYNC
#if __has_builtin(__builtin_amdgcn_s_wait_asynccnt)
#define WAIT_LOADS(pending) do { if (pending) __builtin_amdgcn_s_wait_asynccnt(34); \
                                 else         __builtin_amdgcn_s_wait_asynccnt(0); } while (0)
#else
#define WAIT_LOADS(pending) do { if (pending) asm volatile("s_wait_asynccnt 34" ::: "memory"); \
                                 else         asm volatile("s_wait_asynccnt 0"  ::: "memory"); } while (0)
#endif
#else
#define WAIT_LOADS(pending) do { } while (0)
#endif

// ---------------- WMMA fragment helpers (CDNA5 wave32 layouts) ----------------
__device__ __forceinline__ v16bf frag_A_rm(const __bf16* src, int ld) {
  const int L = threadIdx.x & 31, m = L & 15, hh = L >> 4;
  const __bf16* p = src + m*ld + 8*hh;
  union { uint4 u[2]; v16bf v; } t;
  t.u[0] = *(const uint4*)p;
  t.u[1] = *(const uint4*)(p + 16);
  return t.v;
}
__device__ __forceinline__ v16bf frag_A_cm(const __bf16* src, int ld) {
  const int L = threadIdx.x & 31, m = L & 15, hh = L >> 4;
  v16bf r;
  #pragma unroll
  for (int v = 0; v < 8; ++v) {
    int kb = (v < 4) ? (2*v + 8*hh) : (16 + 2*(v-4) + 8*hh);
    r[2*v]   = src[kb*ld + m];
    r[2*v+1] = src[(kb+1)*ld + m];
  }
  return r;
}
__device__ __forceinline__ v16bf frag_B_cm(const __bf16* src, int ld) {
  const int L = threadIdx.x & 31, n = L & 15, hh = L >> 4;
  const __bf16* p = src + n*ld + 8*hh;
  union { uint4 u[2]; v16bf v; } t;
  t.u[0] = *(const uint4*)p;
  t.u[1] = *(const uint4*)(p + 16);
  return t.v;
}
__device__ __forceinline__ void store_C_T(float* dstT, int ldT, v8f c) {
  const int L = threadIdx.x & 31, n = L & 15, hh = L >> 4;
  union { v8f f; uint4 u[2]; } t; t.f = c;
  float* p = dstT + n*ldT + 8*hh;
  *(uint4*)p       = t.u[0];
  *(uint4*)(p + 4) = t.u[1];
}
__device__ __forceinline__ v8f wmma_bf16(v16bf a, v16bf b, v8f c) {
  return __builtin_amdgcn_wmma_f32_16x16x32_bf16(false, a, false, b, (short)0, c,
                                                 false, false);
}

// ---------------- Kernel A: causal conv1d + SiLU + L2-norm + bf16 pack --------
__global__ __launch_bounds__(128) void conv_norm_kernel(
    const float* __restrict__ xin, const float* __restrict__ cw,
    const float* __restrict__ cb, __bf16* __restrict__ qbf,
    __bf16* __restrict__ kbf, __bf16* __restrict__ vbf)
{
  const int t = blockIdx.x;
  const int hg = blockIdx.y;       // 0..31 q-head, 32..47 k-head, 48..63 v-head
  const int d = threadIdx.x;       // 0..127
  int kind, hloc, cbase;
  if (hg < HQ)            { kind = 0; hloc = hg;          cbase = hg*DK; }
  else if (hg < HQ+HKV)   { kind = 1; hloc = hg - HQ;     cbase = HQ*DK + hloc*DK; }
  else                    { kind = 2; hloc = hg - HQ-HKV; cbase = HQ*DK + HKV*DK + hloc*DV; }
  const int c = cbase + d;

  float y = cb[c];
  #pragma unroll
  for (int j = 0; j < KCONV; ++j) {
    int tt = t - (KCONV - 1) + j;
    if (tt >= 0) y += cw[c*KCONV + j] * xin[(size_t)tt*CONV_DIM + c];
  }
  y = y * (1.f / (1.f + __expf(-y)));   // SiLU

  __shared__ float red[128];
  if (kind < 2) {                        // uniform per-block branch
    red[d] = y * y;
    __syncthreads();
    #pragma unroll
    for (int s = 64; s > 0; s >>= 1) {
      if (d < s) red[d] += red[d + s];
      __syncthreads();
    }
    float scale = rsqrtf(red[0] + 1e-6f);
    if (kind == 0)
      qbf[((size_t)t*HQ + hloc)*DK + d] = (__bf16)(y * scale * 0.08838834764831845f);
    else
      kbf[((size_t)t*HKV + hloc)*DK + d] = (__bf16)(y * scale);
  } else {
    vbf[((size_t)t*HKV + hloc)*DV + d] = (__bf16)y;
  }
}

// ---------------- Kernel B: gating (log-alpha, beta) --------------------------
__global__ __launch_bounds__(256) void gates_kernel(
    const float* __restrict__ a, const float* __restrict__ bvec,
    const float* __restrict__ A_log, const float* __restrict__ dt_bias,
    float* __restrict__ g, float* __restrict__ bet)
{
  int idx = blockIdx.x * 256 + threadIdx.x;
  if (idx < TT*HQ) {
    int h = idx & (HQ - 1);
    float x = a[idx] + dt_bias[h];
    float sp = (x > 20.f) ? x : log1pf(__expf(x));       // softplus
    g[idx]   = -__expf(A_log[h]) * sp;                   // log alpha  (<= 0)
    bet[idx] = 1.f / (1.f + __expf(-bvec[idx]));         // sigmoid
  }
}

// ---------------- Kernel C: chunked delta-rule scan (WMMA + TDM) --------------
__global__ __launch_bounds__(32) void gdn_scan_kernel(
    const __bf16* __restrict__ qbf, const __bf16* __restrict__ kbf,
    const __bf16* __restrict__ vbf, const float* __restrict__ gbuf,
    const float* __restrict__ bbuf, float* __restrict__ out)
{
  const int h    = blockIdx.x >> 3;    // q head
  const int v0   = (blockIdx.x & 7) * DVB;
  const int hk   = h >> 1;             // GQA: HQ/HKV = 2
  const int lane = threadIdx.x;
  const int n    = lane & 15, hh = lane >> 4;

  __shared__ __align__(16) __bf16 kx[2][CHUNK][DK];   // K chunk (double buffer)
  __shared__ __align__(16) __bf16 qx[2][CHUNK][DK];   // Q chunk (double buffer)
  __shared__ __align__(16) __bf16 vxb[2][CHUNK][DVB]; // V slice (double buffer)
  __shared__ __align__(16) float  SfT[DVB][DK];       // state^T fp32
  __shared__ __align__(16) __bf16 SbT[DVB][DK];       // state^T bf16 mirror
  __shared__ __align__(16) float  KKT[CHUNK][CHUNK];  // (K K^T)^T
  __shared__ __align__(16) float  QKT[CHUNK][CHUNK];  // (Q K^T)^T
  __shared__ __align__(16) float  KST[DVB][CHUNK];    // (K S)^T
  __shared__ __align__(16) float  QST[DVB][CHUNK];    // (Q S)^T
  __shared__ __align__(16) __bf16 DltbT[DVB][CHUNK];  // Delta^T
  __shared__ __align__(16) __bf16 DhatT[DVB][CHUNK];  // (decay-to-end * Delta)^T
  __shared__ __align__(16) __bf16 Wo[CHUNK][CHUNK];   // masked q.k * decay
  __shared__ __align__(16) float  DmT[CHUNK][CHUNK];  // DmT[j][i]=exp(cg_i-cg_j), j<=i
  __shared__ float gch[CHUNK], betc[CHUNK], cg[CHUNK], pe[CHUNK];

  for (int i = lane; i < DVB*DK; i += 32) {
    (&SfT[0][0])[i] = 0.f;
    (&SbT[0][0])[i] = (__bf16)0.f;
  }

  // issue chunk loads into buffer `buf` (TDM -> async -> plain fallback)
  auto issue_loads = [&](int t0, int buf) {
#if HAVE_TDM
    tdm_load_2d(lds_byte_off(&qx[buf][0][0]),
                qbf + ((size_t)t0*HQ  + h ) * DK, DK,  CHUNK, HQ*DK);
    tdm_load_2d(lds_byte_off(&kx[buf][0][0]),
                kbf + ((size_t)t0*HKV + hk) * DK, DK,  CHUNK, HKV*DK);
    tdm_load_2d(lds_byte_off(&vxb[buf][0][0]),
                vbf + ((size_t)t0*HKV + hk) * DV + v0, DVB, CHUNK, HKV*DV);
#else
    const __bf16* gq = qbf + ((size_t)(t0+lane)*HQ  + h ) * DK;
    const __bf16* gk = kbf + ((size_t)(t0+lane)*HKV + hk) * DK;
    const __bf16* gv = vbf + ((size_t)(t0+lane)*HKV + hk) * DV + v0;
#if HAVE_ASYNC
    gptr_t gq4 = (gptr_t)(void*)gq;
    gptr_t gk4 = (gptr_t)(void*)gk;
    gptr_t gv4 = (gptr_t)(void*)gv;
    lptr_t lq4 = (lptr_t)(void*)&qx[buf][lane][0];
    lptr_t lk4 = (lptr_t)(void*)&kx[buf][lane][0];
    lptr_t lv4 = (lptr_t)(void*)&vxb[buf][lane][0];
    #pragma unroll
    for (int u = 0; u < (DK*2)/16; ++u) {
      __builtin_amdgcn_global_load_async_to_lds_b128(gq4 + u, lq4 + u, 0, 0);
      __builtin_amdgcn_global_load_async_to_lds_b128(gk4 + u, lk4 + u, 0, 0);
    }
    __builtin_amdgcn_global_load_async_to_lds_b128(gv4,     lv4,     0, 0);
    __builtin_amdgcn_global_load_async_to_lds_b128(gv4 + 1, lv4 + 1, 0, 0);
#else
    const uint4* qs = (const uint4*)gq;
    const uint4* ks = (const uint4*)gk;
    uint4* qd = (uint4*)(&qx[buf][lane][0]);
    uint4* kd = (uint4*)(&kx[buf][lane][0]);
    #pragma unroll
    for (int u = 0; u < (DK*2)/16; ++u) { qd[u] = qs[u]; kd[u] = ks[u]; }
    *(uint4*)(&vxb[buf][lane][0]) = *(const uint4*)gv;
    *(uint4*)(&vxb[buf][lane][8]) = *(const uint4*)(gv + 8);
#endif
#endif
    if (t0 + CHUNK < TT) {   // warm L2 for the chunk after this one
      __builtin_prefetch(qbf + ((size_t)(t0+CHUNK+lane)*HQ  + h ) * DK, 0, 0);
      __builtin_prefetch(kbf + ((size_t)(t0+CHUNK+lane)*HKV + hk) * DK, 0, 0);
    }
  };

  issue_loads(0, 0);

  for (int ch = 0; ch < NCHUNK; ++ch) {
    const int t0  = ch * CHUNK;
    const int cur = ch & 1;
    const __bf16 (*kxc)[DK]  = kx[cur];
    const __bf16 (*qxc)[DK]  = qx[cur];
    const __bf16 (*vxc)[DVB] = vxb[cur];

    const bool more = (ch + 1) < NCHUNK;
    if (more) issue_loads(t0 + CHUNK, cur ^ 1);   // overlap next chunk's DMA
    WAIT_LOADS(more);                              // oldest chunk's loads done

    gch[lane]  = gbuf[(size_t)(t0+lane)*HQ + h];
    betc[lane] = bbuf[(size_t)(t0+lane)*HQ + h];
    __syncthreads();

    // -------- inclusive cumsum of log-decay, per-token exp --------
    {
      float s = 0.f;
      for (int j = 0; j <= lane; ++j) s += gch[j];
      cg[lane] = s;
      pe[lane] = __expf(s);
    }
    __syncthreads();

    // decay matrix DmT[j][i] = exp(cg_i - cg_j) for j<=i, else 0 (causal mask)
    for (int it = 0; it < CHUNK; ++it) {
      int idx = it*32 + lane, j = idx >> 5, i = idx & 31;
      DmT[j][i] = (j <= i) ? __expf(cg[i] - cg[j]) : 0.f;
    }

    // -------- phase 1 WMMA: KK = K K^T, QK = Q K^T, KS = K S, QS = Q S --------
    {
      v16bf aK[2][4], aQ[2][4], bS[4], bK[2][4];
      #pragma unroll
      for (int mt = 0; mt < 2; ++mt)
        #pragma unroll
        for (int kb = 0; kb < DK/32; ++kb) {
          aK[mt][kb] = frag_A_rm(&kxc[mt*16][0] + kb*32, DK);
          aQ[mt][kb] = frag_A_rm(&qxc[mt*16][0] + kb*32, DK);
          bK[mt][kb] = frag_B_cm(&kxc[mt*16][0] + kb*32, DK);
        }
      #pragma unroll
      for (int kb = 0; kb < DK/32; ++kb)
        bS[kb] = frag_B_cm(&SbT[0][kb*32], DK);

      #pragma unroll
      for (int mt = 0; mt < 2; ++mt) {
        v8f accKS = {}; v8f accQS = {};
        #pragma unroll
        for (int kb = 0; kb < DK/32; ++kb) {
          accKS = wmma_bf16(aK[mt][kb], bS[kb], accKS);
          accQS = wmma_bf16(aQ[mt][kb], bS[kb], accQS);
        }
        store_C_T(&KST[0][mt*16], CHUNK, accKS);
        store_C_T(&QST[0][mt*16], CHUNK, accQS);
        #pragma unroll
        for (int nt = 0; nt < 2; ++nt) {
          v8f accKK = {}; v8f accQK = {};
          #pragma unroll
          for (int kb = 0; kb < DK/32; ++kb) {
            accKK = wmma_bf16(aK[mt][kb], bK[nt][kb], accKK);
            accQK = wmma_bf16(aQ[mt][kb], bK[nt][kb], accQK);
          }
          store_C_T(&KKT[nt*16][mt*16], CHUNK, accKK);
          store_C_T(&QKT[nt*16][mt*16], CHUNK, accQK);
        }
      }
    }
    __syncthreads();

    // -------- forward substitution: Delta = (I + M)^{-1} B  (per V-column) ----
    if (lane < DVB) {
      const int v = lane;
      for (int i = 0; i < CHUNK; ++i) {
        float bi  = betc[i];
        float acc = bi * ((float)vxc[i][v] - pe[i] * KST[v][i]);
        for (int j = 0; j < i; ++j)
          acc -= bi * DmT[j][i] * KKT[j][i] * (float)DltbT[v][j];
        DltbT[v][i] = (__bf16)acc;
        DhatT[v][i] = (__bf16)(acc * DmT[i][CHUNK-1]);  // fold decay-to-end
      }
    }
    // Wo[i][j] = (q_i . k_j) * decay, causal-inclusive (DmT already masks j>i)
    for (int it = 0; it < CHUNK; ++it) {
      int idx = it*32 + lane, i = idx >> 5, j = idx & 31;
      Wo[i][j] = (__bf16)(QKT[j][i] * DmT[j][i]);
    }
    __syncthreads();

    // -------- phase 2 WMMA: outputs O = pe*QS + Wo @ Delta --------------------
    {
      v16bf bD = frag_B_cm(&DltbT[0][0], CHUNK);
      #pragma unroll
      for (int mt = 0; mt < 2; ++mt) {
        float* pq = &QST[n][mt*16 + 8*hh];
        union { v8f f; uint4 u[2]; } c;
        c.u[0] = *(uint4*)pq; c.u[1] = *(uint4*)(pq + 4);
        #pragma unroll
        for (int r = 0; r < 8; ++r) c.f[r] *= pe[mt*16 + 8*hh + r];
        v16bf aW = frag_A_rm(&Wo[mt*16][0], CHUNK);
        v8f o = wmma_bf16(aW, bD, c.f);
        #pragma unroll
        for (int r = 0; r < 8; ++r) {
          int row = mt*16 + 8*hh + r;
          out[(size_t)(t0+row)*(HQ*DV) + h*DV + v0 + n] = o[r];
        }
      }
    }

    // -------- state update: S^T = pe_last*S^T + (Khat^T @ Delta)^T ------------
    {
      const float plast = pe[CHUNK-1];
      v16bf bDh = frag_B_cm(&DhatT[0][0], CHUNK);
      for (int dt = 0; dt < DK/16; ++dt) {
        float* ps = &SfT[n][dt*16 + 8*hh];
        union { v8f f; uint4 u[2]; } c;
        c.u[0] = *(uint4*)ps; c.u[1] = *(uint4*)(ps + 4);
        #pragma unroll
        for (int r = 0; r < 8; ++r) c.f[r] *= plast;
        v16bf aKh = frag_A_cm(&kxc[0][dt*16], DK);   // A[m][k] = k_k[dt*16+m]
        v8f sN = wmma_bf16(aKh, bDh, c.f);
        union { v8f f; uint4 u[2]; } o; o.f = sN;
        *(uint4*)ps       = o.u[0];
        *(uint4*)(ps + 4) = o.u[1];
        union { uint4 u; __bf16 b[8]; } sb;
        #pragma unroll
        for (int r = 0; r < 8; ++r) sb.b[r] = (__bf16)sN[r];
        *(uint4*)(&SbT[n][dt*16 + 8*hh]) = sb.u;
      }
    }
    __syncthreads();
  }
}

// ---------------- launch ------------------------------------------------------
extern "C" void kernel_launch(void* const* d_in, const int* in_sizes, int n_in,
                              void* d_out, int out_size, void* d_ws, size_t ws_size,
                              hipStream_t stream) {
  const float* x  = (const float*)d_in[0];
  const float* a  = (const float*)d_in[1];
  const float* b  = (const float*)d_in[2];
  const float* cw = (const float*)d_in[3];
  const float* cb = (const float*)d_in[4];
  const float* Al = (const float*)d_in[5];
  const float* db = (const float*)d_in[6];
  float* out = (float*)d_out;

  char* ws = (char*)d_ws;
  size_t off = 0;
  __bf16* qbf = (__bf16*)(ws + off); off += (size_t)TT*HQ *DK * 2;  // 33.6 MB
  __bf16* kbf = (__bf16*)(ws + off); off += (size_t)TT*HKV*DK * 2;  // 16.8 MB
  __bf16* vbf = (__bf16*)(ws + off); off += (size_t)TT*HKV*DV * 2;  // 16.8 MB
  float*  g   = (float*)(ws + off);  off += (size_t)TT*HQ * 4;
  float*  bet = (float*)(ws + off);

  conv_norm_kernel<<<dim3(TT, HQ + HKV + HKV), 128, 0, stream>>>(x, cw, cb, qbf, kbf, vbf);
  gates_kernel<<<(TT*HQ + 255)/256, 256, 0, stream>>>(a, b, Al, db, g, bet);
  gdn_scan_kernel<<<HQ * (DV/DVB), 32, 0, stream>>>(qbf, kbf, vbf, g, bet, out);
}
